// BRUCell_40209483825953
// MI455X (gfx1250) — matrix operand order repaired and checked by reading, not compiled, not run on hardware
//
#include <hip/hip_runtime.h>

// Problem constants (from reference): B=64, T=2048, D=256, U=256
#define Bk 64
#define Tk 2048
#define Dk 256
#define Uk 256
#define Mk (Bk * Tk)   // 131072 rows

typedef __attribute__((ext_vector_type(4)))  unsigned int v4u;
typedef __attribute__((ext_vector_type(8)))  float        v8f;
typedef __attribute__((ext_vector_type(16))) __bf16       v16bf;

__device__ __forceinline__ unsigned short f2bf(float f) {
  unsigned int u = __float_as_uint(f);
  unsigned int lsb = (u >> 16) & 1u;
  return (unsigned short)((u + 0x7FFFu + lsb) >> 16);   // round-to-nearest-even
}

// ---------------------------------------------------------------------------
// Kernel 0: weight prep. BT[mat][n][k] = bf16(K[mat][k][n]).
// 3 x 256 x 256 = 192K elements; tiny, L2-resident afterwards.
// ---------------------------------------------------------------------------
__global__ __launch_bounds__(256) void bru_prep_weights(
    const float* __restrict__ kz, const float* __restrict__ kr,
    const float* __restrict__ kh, unsigned short* __restrict__ BT) {
  const int n = blockIdx.x;     // output column (0..255)
  const int mat = blockIdx.y;   // 0..2
  const int k = threadIdx.x;    // 0..255
  const float* K = (mat == 0) ? kz : ((mat == 1) ? kr : kh);
  BT[((size_t)mat * Uk + n) * Dk + k] = f2bf(K[(size_t)k * Uk + n]);
}

// ---------------------------------------------------------------------------
// Kernel 1: fused triple GEMM  P[mat] = x @ K[mat]  (bf16 WMMA, f32 accum)
// Block: 256 threads (8 waves), owns 32 rows of x staged in LDS as bf16.
// 24 wave-tasks = {2 m-subtiles x 4 n-groups(64) x 3 matrices}; each task:
// 4 accumulators of 16x16, K looped in steps of 32 -> 32 WMMAs/task.
// A fragments reload from LDS each K step (reused by 4 WMMAs); B streams from L2.
// ---------------------------------------------------------------------------
__global__ __launch_bounds__(256) void bru_gemm(
    const float* __restrict__ x, const unsigned short* __restrict__ BT,
    float* __restrict__ P0, float* __restrict__ P1, float* __restrict__ P2) {
  // +8 halves of row padding: row stride 528B = 132 dwords == 4 (mod 64 banks)
  __shared__ __align__(16) unsigned short As[32][Dk + 8];

  const int tid = threadIdx.x;
  const size_t m0 = (size_t)blockIdx.x * 32;

  // Stage 32 rows of x, converting fp32 -> bf16 (coalesced 32 elems/thread).
  for (int i = tid; i < 32 * Dk; i += 256) {
    const int r = i >> 8, c = i & (Dk - 1);
    As[r][c] = f2bf(x[(m0 + r) * Dk + c]);
  }
  __syncthreads();

  const int wave = tid >> 5;
  const int lane = tid & 31;
  const int lrow = lane & 15;   // A: row within tile, B: column, C/D: column
  const int lhi  = lane >> 4;   // half-wave selector

  for (int task = wave; task < 24; task += 8) {     // uniform per wave
    const int msub = task & 1;
    const int ngrp = (task >> 1) & 3;
    const int mat  = task >> 3;
    const unsigned short* Bm = BT + (size_t)mat * Uk * Dk;

    const v8f vzero = {0.f, 0.f, 0.f, 0.f, 0.f, 0.f, 0.f, 0.f};
    v8f acc[4] = {vzero, vzero, vzero, vzero};

    for (int kk = 0; kk < Dk; kk += 32) {
      // A fragment (16x32): V0-3 <- K [kk + lhi*8 .. +7], V4-7 <- +16
      union { v4u u[2]; v16bf v; } ua;
      const unsigned short* arow = &As[msub * 16 + lrow][0];
      ua.u[0] = *reinterpret_cast<const v4u*>(arow + kk + lhi * 8);
      ua.u[1] = *reinterpret_cast<const v4u*>(arow + kk + 16 + lhi * 8);

      for (int j = 0; j < 4; ++j) {
        // B fragment (32x16): lane holds column lrow, K [kk + lhi*8 .. +7]
        // and +16 (BT is pre-transposed so K is contiguous).
        union { v4u u[2]; v16bf v; } ub;
        const unsigned short* brow =
            Bm + (size_t)(ngrp * 64 + j * 16 + lrow) * Dk + kk + lhi * 8;
        ub.u[0] = *reinterpret_cast<const v4u*>(brow);
        ub.u[1] = *reinterpret_cast<const v4u*>(brow + 16);

        acc[j] = __builtin_amdgcn_wmma_f32_16x16x32_bf16(
            /*neg_a=*/false, ua.v, /*neg_b=*/false, ub.v,
            /*c_mod=*/(short)0, acc[j], /*reuse_a=*/false, /*reuse_b=*/false);
      }
    }

    // Epilogue: C/D layout — VGPR v holds M = v + lhi*8, N = lrow.
    // P is written once and read once (streaming) -> non-temporal stores.
    float* P = (mat == 0) ? P0 : ((mat == 1) ? P1 : P2);
    const size_t row0 = m0 + msub * 16 + lhi * 8;
    for (int j = 0; j < 4; ++j) {
      const int ncol = ngrp * 64 + j * 16 + lrow;
      for (int v = 0; v < 8; ++v)
        __builtin_nontemporal_store(acc[j][v], &P[(row0 + v) * Uk + ncol]);
    }
  }
}

// ---------------------------------------------------------------------------
// Kernel 2: elementwise time scan. One thread per (b, u); coalesced across u.
// The projection loads for step t do NOT depend on h, so we software-pipeline
// one step ahead in registers on top of an 8-step global_prefetch. Projection
// reads are single-use streaming -> non-temporal loads; same for out stores.
// tanh/sigmoid via hardware v_exp_f32 in saturation-safe form.
// ---------------------------------------------------------------------------
__global__ __launch_bounds__(256) void bru_scan(
    const float* __restrict__ xz, const float* __restrict__ xr,
    const float* __restrict__ xh,
    const float* __restrict__ mz, const float* __restrict__ mr,
    const float* __restrict__ bz, const float* __restrict__ br,
    const float* __restrict__ bh, float* __restrict__ out) {
  const int b = blockIdx.x;
  const int u = threadIdx.x;
  const float mzv = mz[u], mrv = mr[u];
  const float bzv = bz[u], brv = br[u], bhv = bh[u];

  float h = 0.0f;
  size_t idx = (size_t)b * Tk * Uk + u;

  // Pipeline prologue: step-0 operands.
  float nz = __builtin_nontemporal_load(&xz[idx]);
  float nr = __builtin_nontemporal_load(&xr[idx]);
  float nh = __builtin_nontemporal_load(&xh[idx]);

  for (int t = 0; t < Tk; ++t, idx += Uk) {
    const float az = nz + bzv;
    const float ar = nr + brv;
    const float ah = nh + bhv;

    // Issue next step's loads (independent of h) before the serial math.
    if (t + 1 < Tk) {
      nz = __builtin_nontemporal_load(&xz[idx + Uk]);
      nr = __builtin_nontemporal_load(&xr[idx + Uk]);
      nh = __builtin_nontemporal_load(&xh[idx + Uk]);
      if (t + 8 < Tk) {
        __builtin_prefetch(&xz[idx + 8 * Uk], 0, 0);
        __builtin_prefetch(&xr[idx + 8 * Uk], 0, 0);
        __builtin_prefetch(&xh[idx + 8 * Uk], 0, 0);
      }
    }

    // r = tanh(ar + h*mr) + 1 = 2 - 2/(exp(2x)+1)   (saturation-safe)
    const float e2r = __expf(2.0f * (ar + h * mrv));
    const float r = 2.0f - 2.0f / (e2r + 1.0f);
    // z = sigmoid(az + h*mz)
    const float z = 1.0f / (1.0f + __expf(-(az + h * mzv)));
    // hh = tanh(ah + r*h) = 1 - 2/(exp(2x)+1)
    const float e2h = __expf(2.0f * (ah + r * h));
    const float hh = 1.0f - 2.0f / (e2h + 1.0f);

    h = (1.0f - z) * hh + z * h;
    __builtin_nontemporal_store(h, &out[idx]);
  }
}

// ---------------------------------------------------------------------------
extern "C" void kernel_launch(void* const* d_in, const int* in_sizes, int n_in,
                              void* d_out, int out_size, void* d_ws, size_t ws_size,
                              hipStream_t stream) {
  const float* x  = (const float*)d_in[0];
  const float* kz = (const float*)d_in[1];
  const float* kr = (const float*)d_in[2];
  const float* kh = (const float*)d_in[3];
  const float* mz = (const float*)d_in[4];
  const float* mr = (const float*)d_in[5];
  const float* bz = (const float*)d_in[6];
  const float* br = (const float*)d_in[7];
  const float* bh = (const float*)d_in[8];

  // Workspace layout: [0, 384KB) transposed bf16 weights; [1MB, ...) three
  // fp32 projection buffers of M*U floats each (128 MB apiece).
  unsigned short* BT = (unsigned short*)d_ws;
  float* P0 = (float*)((char*)d_ws + (1u << 20));
  float* P1 = P0 + (size_t)Mk * Uk;
  float* P2 = P1 + (size_t)Mk * Uk;

  bru_prep_weights<<<dim3(Uk, 3), Dk, 0, stream>>>(kz, kr, kh, BT);
  bru_gemm<<<Mk / 32, 256, 0, stream>>>(x, BT, P0, P1, P2);
  bru_scan<<<Bk, Uk, 0, stream>>>(P0, P1, P2, mz, mr, bz, br, bh, (float*)d_out);
}
